// MambaBlock_41334765256723
// MI455X (gfx1250) — compile-verified, matrix operand order checked by reading, not run on hardware
//
#include <hip/hip_runtime.h>
#include <hip/hip_bf16.h>

// ---------------------------------------------------------------------------
// Mamba block for MI455X (gfx1250, wave32, WMMA + Tensor Data Mover).
// Pipeline: LayerNorm -> in_proj (TDM+LDS WMMA) -> causal conv + SiLU ->
//           x_proj (WMMA) -> dt_proj (WMMA, softplus) ->
//           selective scan (reg-resident 16-state recurrence) ->
//           out_proj (TDM+LDS WMMA, residual epilogue).
// ---------------------------------------------------------------------------

#define D_MODEL 1024
#define D_INNER 2048
#define D_STATE 16
#define D_CONV  4
#define DT_RANK 64
#define B_SZ    2
#define SEQ_L   2048
#define M_TOK   (B_SZ * SEQ_L)      // 4096 token rows

typedef __attribute__((ext_vector_type(16))) __bf16    v16bf;
typedef __attribute__((ext_vector_type(8)))  __bf16    v8bf;
typedef __attribute__((ext_vector_type(8)))  float     v8f;
typedef __attribute__((ext_vector_type(4)))  unsigned  v4u;
typedef __attribute__((ext_vector_type(8)))  int       v8i;
typedef __attribute__((ext_vector_type(4)))  int       v4i;

static __device__ __forceinline__ __bf16 f2bf(float f) {
    unsigned u = __builtin_bit_cast(unsigned, f);
    unsigned r = (u + 0x7FFFu + ((u >> 16) & 1u)) >> 16;
    return __builtin_bit_cast(__bf16, (unsigned short)r);
}

// low 32 bits of a flat LDS pointer == byte offset within the wave's LDS
static __device__ __forceinline__ unsigned lds_offset(const void* p) {
    return (unsigned)(unsigned long long)(size_t)p;
}

// -------------------- Tensor Data Mover 2D tile load -----------------------
// DMA a (tile_rows x tile_cols) bf16 tile (row stride = row_stride elements)
// from global memory into LDS at lds_off. Wave-level op, tracked by TENSORcnt.
static __device__ __forceinline__ void tdm_load_tile_bf16(
    const __bf16* gptr, unsigned lds_off,
    unsigned tile_rows, unsigned tile_cols,
    unsigned long long tensor_rows, unsigned long long tensor_cols,
    unsigned long long row_stride) {
    unsigned long long ga = (unsigned long long)(size_t)gptr;
    v4u g0;
    g0[0] = 1u;                                        // count=1, user mode
    g0[1] = lds_off;                                   // LDS byte address
    g0[2] = (unsigned)(ga & 0xFFFFFFFFu);              // global_addr[31:0]
    g0[3] = (unsigned)((ga >> 32) & 0x01FFFFFFu)       // global_addr[56:32]
          | (2u << 30);                                // type = 2 (image)
    v8i g1;
    g1[0] = (int)(1u << 16);                           // data_size=1 (2 bytes)
    g1[1] = (int)((tensor_cols & 0xFFFFu) << 16);      // tensor_dim0[15:0]
    g1[2] = (int)(((tensor_cols >> 16) & 0xFFFFu)      // tensor_dim0[31:16]
          | ((tensor_rows & 0xFFFFu) << 16));          // tensor_dim1[15:0]
    g1[3] = (int)(((tensor_rows >> 16) & 0xFFFFu)      // tensor_dim1[31:16]
          | (tile_cols << 16));                        // tile_dim0
    g1[4] = (int)tile_rows;                            // tile_dim1 (tile_dim2=0)
    g1[5] = (int)(row_stride & 0xFFFFFFFFu);           // tensor_dim0_stride lo
    g1[6] = (int)((row_stride >> 32) & 0xFFFFu);       // stride hi; dim1_stride=0
    g1[7] = 0;
    v4i z4 = {0, 0, 0, 0};                             // groups 2/3 unused (2D)
    v8i z8 = {0, 0, 0, 0, 0, 0, 0, 0};                 // trailing group (unused)
    __builtin_amdgcn_tensor_load_to_lds(g0, g1, z4, z4, z8, 0);
}

// ------------------------------- LayerNorm ---------------------------------
__global__ void ln_kernel(const float* __restrict__ hs,
                          const float* __restrict__ g,
                          const float* __restrict__ b,
                          __bf16* __restrict__ out) {
    const int row = blockIdx.x;
    const int tid = threadIdx.x;
    const size_t base = (size_t)row * D_MODEL;
    __shared__ float rs[256];
    __shared__ float rq[256];
    float s = 0.f, q = 0.f;
    for (int i = tid; i < D_MODEL; i += 256) {
        float v = hs[base + i];
        s += v; q += v * v;
    }
    rs[tid] = s; rq[tid] = q;
    __syncthreads();
    for (int off = 128; off > 0; off >>= 1) {
        if (tid < off) { rs[tid] += rs[tid + off]; rq[tid] += rq[tid + off]; }
        __syncthreads();
    }
    const float mean = rs[0] * (1.0f / D_MODEL);
    const float var  = rq[0] * (1.0f / D_MODEL) - mean * mean;
    const float rstd = rsqrtf(var + 1e-5f);
    for (int i = tid; i < D_MODEL; i += 256) {
        float v = (hs[base + i] - mean) * rstd * g[i] + b[i];
        out[base + i] = f2bf(v);
    }
}

// ------------------------ fp32 -> bf16 conversions -------------------------
__global__ void cvt_flat(const float* __restrict__ in, __bf16* __restrict__ out, long n) {
    long i = (long)blockIdx.x * blockDim.x + threadIdx.x;
    if (i < n) out[i] = f2bf(in[i]);
}

__global__ void cvt_rows(const float* __restrict__ in, __bf16* __restrict__ out,
                         int rows, int cin, int ctake) {
    long i = (long)blockIdx.x * blockDim.x + threadIdx.x;
    long n = (long)rows * ctake;
    if (i >= n) return;
    int r = (int)(i / ctake), c = (int)(i % ctake);
    out[i] = f2bf(in[(size_t)r * cin + c]);
}

// ----------------- direct-from-global WMMA GEMM (small cases) --------------
// C[M,N] = A[M,K] * B[N,K]^T. One wave per 16x16 tile; 4 waves per block.
// EPI: 0 = plain f32 store, 1 = softplus(acc + aux[n]), 2 = acc + aux[m*N+n].
template <int EPI>
__global__ void gemm_bf16_wmma(const __bf16* __restrict__ A,
                               const __bf16* __restrict__ Bw,
                               float* __restrict__ C,
                               int M, int N, int K,
                               const float* __restrict__ aux) {
    const int lane  = threadIdx.x & 31;
    const int wid   = threadIdx.x >> 5;
    const int nbase = (blockIdx.x * 4 + wid) * 16;
    if (nbase >= N) return;                 // uniform per wave
    const int mbase = blockIdx.y * 16;
    const int half  = lane >> 4;
    const int l15   = lane & 15;

    const size_t arow = (size_t)(mbase + l15) * K + (size_t)half * 8;
    const size_t brow = (size_t)(nbase + l15) * K + (size_t)half * 8;

    v8f acc = {};
    for (int k0 = 0; k0 < K; k0 += 32) {
        v8bf alo = *(const v8bf*)(A  + arow + k0);
        v8bf ahi = *(const v8bf*)(A  + arow + k0 + 16);
        v8bf blo = *(const v8bf*)(Bw + brow + k0);
        v8bf bhi = *(const v8bf*)(Bw + brow + k0 + 16);
        __builtin_prefetch(A  + arow + k0 + 64, 0, 1);
        __builtin_prefetch(Bw + brow + k0 + 64, 0, 1);
        v16bf av, bv;
#pragma unroll
        for (int i = 0; i < 8; ++i) {
            av[i] = alo[i]; av[i + 8] = ahi[i];
            bv[i] = blo[i]; bv[i + 8] = bhi[i];
        }
        acc = __builtin_amdgcn_wmma_f32_16x16x32_bf16(
            false, av, false, bv, (short)0, acc, false, false);
    }
    const int n = nbase + l15;
#pragma unroll
    for (int v = 0; v < 8; ++v) {
        int m = mbase + half * 8 + v;
        float val = acc[v];
        if (EPI == 1) {
            float x = val + aux[n];
            val = (x > 20.f) ? x : log1pf(__expf(x));
        } else if (EPI == 2) {
            val += aux[(size_t)m * N + n];
        }
        C[(size_t)m * N + n] = val;
    }
}

// ------------- TDM + LDS double-buffered WMMA GEMM (big cases) -------------
// Block = 256 thr = 8 waves (2 Mx4 N). Block tile 64(M) x 128(N), BK=32.
// TDM DMAs A(64x32) and B(128x32) bf16 tiles into LDS; waves read fragments
// with ds_load_b128 and do 4 WMMAs each per K-step.
#define BM 64
#define BN 128
#define BK 32

template <int EPI>
__global__ void gemm_bf16_wmma_lds(const __bf16* __restrict__ A,
                                   const __bf16* __restrict__ Bw,
                                   float* __restrict__ C,
                                   int M, int N, int K,
                                   const float* __restrict__ aux) {
    __shared__ __bf16 Abuf[2][BM * BK];   // 2 x 4 KB
    __shared__ __bf16 Bbuf[2][BN * BK];   // 2 x 8 KB

    const int lane = threadIdx.x & 31;
    const int wid  = threadIdx.x >> 5;
    const int wm   = wid >> 2;            // 0..1  (32-row strip)
    const int wn   = wid & 3;             // 0..3  (32-col strip)
    const int half = lane >> 4;
    const int l15  = lane & 15;

    const int mbase = blockIdx.y * BM;
    const int nbase = blockIdx.x * BN;

    v8f acc[2][2] = {};

    // prologue: stage K-step 0 into buffer 0 (one wave issues the DMA)
    if (wid == 0) {
        tdm_load_tile_bf16(A  + (size_t)mbase * K, lds_offset(&Abuf[0][0]),
                           BM, BK, (unsigned long long)M, (unsigned long long)K,
                           (unsigned long long)K);
        tdm_load_tile_bf16(Bw + (size_t)nbase * K, lds_offset(&Bbuf[0][0]),
                           BN, BK, (unsigned long long)N, (unsigned long long)K,
                           (unsigned long long)K);
    }

    for (int k0 = 0; k0 < K; k0 += BK) {
        const int cur = (k0 / BK) & 1;
        __builtin_amdgcn_s_wait_tensorcnt(0);   // no-op for waves with cnt==0
        __syncthreads();                        // tile `cur` visible to all;
                                                // also: all reads of `cur^1` done
        if (wid == 0 && k0 + BK < K) {
            tdm_load_tile_bf16(A  + (size_t)mbase * K + k0 + BK,
                               lds_offset(&Abuf[cur ^ 1][0]),
                               BM, BK, (unsigned long long)M, (unsigned long long)K,
                               (unsigned long long)K);
            tdm_load_tile_bf16(Bw + (size_t)nbase * K + k0 + BK,
                               lds_offset(&Bbuf[cur ^ 1][0]),
                               BN, BK, (unsigned long long)N, (unsigned long long)K,
                               (unsigned long long)K);
        }

        v16bf afrag[2], bfrag[2];
#pragma unroll
        for (int mt = 0; mt < 2; ++mt) {
            const int r = wm * 32 + mt * 16 + l15;
            const v8bf* pa = (const v8bf*)&Abuf[cur][r * BK + half * 8];
            v8bf lo = pa[0], hi = pa[2];
#pragma unroll
            for (int i = 0; i < 8; ++i) { afrag[mt][i] = lo[i]; afrag[mt][i + 8] = hi[i]; }
        }
#pragma unroll
        for (int nt = 0; nt < 2; ++nt) {
            const int r = wn * 32 + nt * 16 + l15;
            const v8bf* pb = (const v8bf*)&Bbuf[cur][r * BK + half * 8];
            v8bf lo = pb[0], hi = pb[2];
#pragma unroll
            for (int i = 0; i < 8; ++i) { bfrag[nt][i] = lo[i]; bfrag[nt][i + 8] = hi[i]; }
        }
#pragma unroll
        for (int mt = 0; mt < 2; ++mt)
#pragma unroll
            for (int nt = 0; nt < 2; ++nt)
                acc[mt][nt] = __builtin_amdgcn_wmma_f32_16x16x32_bf16(
                    false, afrag[mt], false, bfrag[nt],
                    (short)0, acc[mt][nt], false, false);
    }

#pragma unroll
    for (int mt = 0; mt < 2; ++mt) {
#pragma unroll
        for (int nt = 0; nt < 2; ++nt) {
            const int n = nbase + wn * 32 + nt * 16 + l15;
#pragma unroll
            for (int v = 0; v < 8; ++v) {
                int m = mbase + wm * 32 + mt * 16 + half * 8 + v;
                float val = acc[mt][nt][v];
                if (EPI == 1) {
                    float x = val + aux[n];
                    val = (x > 20.f) ? x : log1pf(__expf(x));
                } else if (EPI == 2) {
                    val += aux[(size_t)m * N + n];
                }
                C[(size_t)m * N + n] = val;
            }
        }
    }
}

// ------------------- depthwise causal conv1d + SiLU ------------------------
__global__ void conv_silu_kernel(const float* __restrict__ xz,
                                 const float* __restrict__ cw,
                                 const float* __restrict__ cb,
                                 float* __restrict__ xconv,
                                 __bf16* __restrict__ xconv_bf) {
    long idx = (long)blockIdx.x * blockDim.x + threadIdx.x;
    if (idx >= (long)M_TOK * D_INNER) return;
    int row = (int)(idx / D_INNER);
    int c   = (int)(idx % D_INNER);
    int bb  = row / SEQ_L;
    int t   = row % SEQ_L;
    float sum = cb[c];
#pragma unroll
    for (int j = 0; j < D_CONV; ++j) {
        int tt = t - (D_CONV - 1) + j;
        if (tt >= 0)
            sum += cw[c * D_CONV + j] * xz[(size_t)(bb * SEQ_L + tt) * (2 * D_INNER) + c];
    }
    float v = sum / (1.f + __expf(-sum));
    xconv[idx]    = v;
    xconv_bf[idx] = f2bf(v);
}

// --------------------------- selective scan --------------------------------
__global__ void scan_kernel(const float* __restrict__ dtf,
                            const float* __restrict__ xdbl,
                            const float* __restrict__ xconv,
                            const float* __restrict__ xz,
                            const float* __restrict__ A_log,
                            const float* __restrict__ Dp,
                            __bf16* __restrict__ ybf) {
    int ch = blockIdx.x * blockDim.x + threadIdx.x;
    if (ch >= B_SZ * D_INNER) return;
    int bb = ch / D_INNER;
    int c  = ch % D_INNER;

    float Av[D_STATE];
#pragma unroll
    for (int s = 0; s < D_STATE; ++s) Av[s] = -__expf(A_log[(size_t)c * D_STATE + s]);
    const float Dc = Dp[c];

    float hst[D_STATE];
#pragma unroll
    for (int s = 0; s < D_STATE; ++s) hst[s] = 0.f;

    for (int t = 0; t < SEQ_L; ++t) {
        const size_t row = (size_t)bb * SEQ_L + t;
        const float dt = dtf[row * D_INNER + c];
        const float xv = xconv[row * D_INNER + c];
        const float zv = xz[row * (2 * D_INNER) + D_INNER + c];
        const float* bc = xdbl + row * 96 + DT_RANK;
        float y = 0.f;
#pragma unroll
        for (int s = 0; s < D_STATE; ++s) {
            float dA  = __expf(dt * Av[s]);
            float dBu = dt * bc[s] * xv;
            hst[s] = dA * hst[s] + dBu;
            y += hst[s] * bc[D_STATE + s];
        }
        y += Dc * xv;
        y *= zv / (1.f + __expf(-zv));
        ybf[row * D_INNER + c] = f2bf(y);
    }
}

// ------------------------------- launcher ----------------------------------
static inline size_t align256(size_t x) { return (x + 255) & ~(size_t)255; }

extern "C" void kernel_launch(void* const* d_in, const int* in_sizes, int n_in,
                              void* d_out, int out_size, void* d_ws, size_t ws_size,
                              hipStream_t stream) {
    const float* hs    = (const float*)d_in[0];
    const float* ln_g  = (const float*)d_in[1];
    const float* ln_b  = (const float*)d_in[2];
    const float* w_in  = (const float*)d_in[3];
    const float* cw    = (const float*)d_in[4];
    const float* cb    = (const float*)d_in[5];
    const float* w_x   = (const float*)d_in[6];
    const float* w_dt  = (const float*)d_in[7];
    const float* b_dt  = (const float*)d_in[8];
    const float* A_log = (const float*)d_in[9];
    const float* Dp    = (const float*)d_in[10];
    const float* w_out = (const float*)d_in[11];
    float* out = (float*)d_out;

    char* p = (char*)d_ws;
    size_t off = 0;
    auto take = [&](size_t bytes) { char* r = p + off; off = align256(off + bytes); return r; };

    __bf16* h_bf     = (__bf16*)take((size_t)M_TOK * D_MODEL * 2);
    __bf16* w1_bf    = (__bf16*)take((size_t)2 * D_INNER * D_MODEL * 2);
    __bf16* w2_bf    = (__bf16*)take((size_t)96 * D_INNER * 2);
    __bf16* w3_bf    = (__bf16*)take((size_t)D_INNER * DT_RANK * 2);
    __bf16* w4_bf    = (__bf16*)take((size_t)D_MODEL * D_INNER * 2);
    float*  xz       = (float*) take((size_t)M_TOK * 2 * D_INNER * 4);
    float*  xconv    = (float*) take((size_t)M_TOK * D_INNER * 4);
    __bf16* xconv_bf = (__bf16*)take((size_t)M_TOK * D_INNER * 2);
    float*  x_dbl    = (float*) take((size_t)M_TOK * 96 * 4);
    __bf16* dt_bf    = (__bf16*)take((size_t)M_TOK * DT_RANK * 2);
    float*  dtf      = (float*) take((size_t)M_TOK * D_INNER * 4);
    __bf16* ybf      = (__bf16*)take((size_t)M_TOK * D_INNER * 2);
    (void)ws_size; (void)in_sizes; (void)n_in; (void)out_size;

    // 1) LayerNorm -> bf16
    ln_kernel<<<M_TOK, 256, 0, stream>>>(hs, ln_g, ln_b, h_bf);

    // 2) weights -> bf16
    {
        long n1 = (long)2 * D_INNER * D_MODEL;
        cvt_flat<<<(unsigned)((n1 + 255) / 256), 256, 0, stream>>>(w_in, w1_bf, n1);
        long n2 = (long)96 * D_INNER;
        cvt_flat<<<(unsigned)((n2 + 255) / 256), 256, 0, stream>>>(w_x, w2_bf, n2);
        long n3 = (long)D_INNER * DT_RANK;
        cvt_flat<<<(unsigned)((n3 + 255) / 256), 256, 0, stream>>>(w_dt, w3_bf, n3);
        long n4 = (long)D_MODEL * D_INNER;
        cvt_flat<<<(unsigned)((n4 + 255) / 256), 256, 0, stream>>>(w_out, w4_bf, n4);
    }

    // 3) in_proj (TDM+LDS): xz[M,4096] = h @ w_in^T
    gemm_bf16_wmma_lds<0><<<dim3((2 * D_INNER) / BN, M_TOK / BM), 256, 0, stream>>>(
        h_bf, w1_bf, xz, M_TOK, 2 * D_INNER, D_MODEL, nullptr);

    // 4) depthwise causal conv + SiLU
    {
        long n = (long)M_TOK * D_INNER;
        conv_silu_kernel<<<(unsigned)((n + 255) / 256), 256, 0, stream>>>(
            xz, cw, cb, xconv, xconv_bf);
    }

    // 5) x_proj: x_dbl[M,96] = xconv @ w_x^T
    gemm_bf16_wmma<0><<<dim3((96 + 63) / 64, M_TOK / 16), 128, 0, stream>>>(
        xconv_bf, w2_bf, x_dbl, M_TOK, 96, D_INNER, nullptr);

    // 6) dt slice -> bf16
    {
        long n = (long)M_TOK * DT_RANK;
        cvt_rows<<<(unsigned)((n + 255) / 256), 256, 0, stream>>>(
            x_dbl, dt_bf, M_TOK, 96, DT_RANK);
    }

    // 7) dt_proj + softplus
    gemm_bf16_wmma<1><<<dim3((D_INNER + 63) / 64, M_TOK / 16), 128, 0, stream>>>(
        dt_bf, w3_bf, dtf, M_TOK, D_INNER, DT_RANK, b_dt);

    // 8) selective scan (fused +D*x and *silu(z))
    scan_kernel<<<(B_SZ * D_INNER) / 256, 256, 0, stream>>>(
        dtf, x_dbl, xconv, xz, A_log, Dp, ybf);

    // 9) out_proj + residual (TDM+LDS)
    gemm_bf16_wmma_lds<2><<<dim3(D_MODEL / BN, M_TOK / BM), 256, 0, stream>>>(
        ybf, w4_bf, out, M_TOK, D_MODEL, D_INNER, hs);
}
